// MaskAwareConvLSTM_18339510354417
// MI455X (gfx1250) — matrix operand order repaired and checked by reading, not compiled, not run on hardware
//
#include <hip/hip_runtime.h>
#include <hip/hip_bf16.h>
#include <math.h>

// ---------------------------------------------------------------------------
// ConvLSTM2D forward (Keras semantics, return_sequences=False) on gfx1250.
// Implicit-GEMM per timestep via V_WMMA_F32_16X16X32_F16, fp32 accumulate.
//   B=8 T=16 H=W=96 C=8 F=64 ; N = 4F = 256 ; K = 9*8(+pad 24) + 9*64 = 672
// ---------------------------------------------------------------------------

typedef _Float16 v16h __attribute__((ext_vector_type(16)));
typedef _Float16 v8h  __attribute__((ext_vector_type(8)));
typedef float    v8f  __attribute__((ext_vector_type(8)));
typedef float    v4f  __attribute__((ext_vector_type(4)));

#define BATCH   8
#define TSTEPS  16
#define HH      96
#define WW      96
#define CIN     8
#define FF      64
#define NGATE   256          // 4*FF
#define KX      72           // 9*CIN
#define KXPAD   96           // padded to 3 chunks of 32
#define KH      576          // 9*FF
#define KTOT    672          // KXPAD + KH
#define NCHUNK  21           // KTOT/32
#define PIX_PER_BLK 32       // M tile
#define TILES_PER_ROW 3      // WW / PIX_PER_BLK
#define BLOCKS_PER_STEP (BATCH * HH * TILES_PER_ROW)   // 2304
#define HSIZE   (BATCH * HH * WW * FF)                  // 4718592 elems

__device__ __forceinline__ float hsig(float z) {
    return fminf(fmaxf(0.2f * z + 0.5f, 0.0f), 1.0f);
}

// ---------------------------------------------------------------------------
// Pack Wx (3,3,8,256) + Wh (3,3,64,256) fp32 -> f16 B-fragments in
// per-lane order (contiguous k, same as A side):  flat = ((kc*16 + nsub)*32 + lane)*16 + e
// with k = kc*32 + (lane>>4)*16 + e ,  n = nsub*16 + (lane&15).
// K rows 72..95 are zero padding between the x-part and the h-part.
// ---------------------------------------------------------------------------
__global__ __launch_bounds__(256) void pack_weights_kernel(
    const float* __restrict__ Wx, const float* __restrict__ Wh,
    _Float16* __restrict__ wpack)
{
    int flat = blockIdx.x * 256 + threadIdx.x;
    if (flat >= NCHUNK * 16 * 32 * 16) return;
    int e    = flat & 15;
    int lane = (flat >> 4) & 31;
    int ns   = (flat >> 9) & 15;
    int kc   = flat >> 13;
    int k = kc * 32 + (lane >> 4) * 16 + e;
    int n = ns * 16 + (lane & 15);
    float v = 0.0f;
    if (k < KX)            v = Wx[k * NGATE + n];            // (tap*8+c)*256+n == k*256+n
    else if (k >= KXPAD)   v = Wh[(k - KXPAD) * NGATE + n];  // (tap*64+f)*256+n
    wpack[flat] = (_Float16)v;
}

__global__ __launch_bounds__(256) void zero_h_kernel(_Float16* __restrict__ h0)
{
    int i = blockIdx.x * 256 + threadIdx.x;
    if (i < HSIZE) h0[i] = (_Float16)0.0f;
}

// ---------------------------------------------------------------------------
// One timestep: z = conv(x_t,Wx)+conv(h,Wh)+b ; gates ; update c,h.
// Block: 256 threads (8 waves) -> 32 pixels x 256 gate channels.
// ---------------------------------------------------------------------------
__global__ __launch_bounds__(256) void convlstm_step_kernel(
    const float*    __restrict__ x,        // (B,T,H,W,C) fp32
    const _Float16* __restrict__ wpack,    // packed B fragments
    const float*    __restrict__ bias,     // (256,)
    const _Float16* __restrict__ h_in,     // (B,H,W,F) f16
    _Float16*       __restrict__ h_out,    // (B,H,W,F) f16
    float*          __restrict__ c_state,  // (B,H,W,F) fp32, in-place
    float*          __restrict__ out_f32,  // d_out on last step, else null
    int t)
{
    __shared__ __align__(16) _Float16 As[PIX_PER_BLK * KTOT];  // 43008 B im2col
    float* zbuf = reinterpret_cast<float*>(As);                // aliased after GEMM (32 KB)

    const int tid = threadIdx.x;
    const int g   = blockIdx.x;
    const int b   = g / (HH * TILES_PER_ROW);
    const int rem = g % (HH * TILES_PER_ROW);
    const int y   = rem / TILES_PER_ROW;
    const int x0  = (rem % TILES_PER_ROW) * PIX_PER_BLK;

    // ---- stage im2col rows into LDS (zero halo at borders), vectorized ----
    // x-part: one (pixel, tap) = 8 contiguous fp32 channels -> 8 f16 (16 B)
    const float* xt = x + (size_t)(b * TSTEPS + t) * (HH * WW * CIN);
    for (int idx = tid; idx < PIX_PER_BLK * 9; idx += 256) {
        int p = idx / 9, tap = idx % 9;
        int gy = y + tap / 3 - 1;
        int gx = x0 + p + tap % 3 - 1;
        v8h out = {};
        if ((unsigned)gy < (unsigned)HH && (unsigned)gx < (unsigned)WW) {
            const v4f* src = reinterpret_cast<const v4f*>(&xt[(gy * WW + gx) * CIN]);
            v4f f0 = src[0], f1 = src[1];
            #pragma unroll
            for (int e = 0; e < 4; ++e) { out[e] = (_Float16)f0[e]; out[4 + e] = (_Float16)f1[e]; }
        }
        *reinterpret_cast<v8h*>(&As[p * KTOT + tap * 8]) = out;   // byte off p*1344+tap*16
    }
    // K-padding rows 72..95: three 16-B zero stores per pixel
    for (int idx = tid; idx < PIX_PER_BLK * 3; idx += 256) {
        int p = idx / 3, j = idx % 3;
        v8h z = {};
        *reinterpret_cast<v8h*>(&As[p * KTOT + KX + j * 8]) = z;  // byte off p*1344+144+j*16
    }
    // h-part: one (pixel, tap, f8) = 8 contiguous f16 filters (16 B load + 16 B LDS store)
    const _Float16* hb = h_in + (size_t)b * (HH * WW * FF);
    for (int idx = tid; idx < PIX_PER_BLK * (KH / 8); idx += 256) {
        int p = idx / (KH / 8);
        int r = idx % (KH / 8);          // 0..71
        int tap = r >> 3, f8 = r & 7;
        int gy = y + tap / 3 - 1;
        int gx = x0 + p + tap % 3 - 1;
        v8h v = {};
        if ((unsigned)gy < (unsigned)HH && (unsigned)gx < (unsigned)WW)
            v = *reinterpret_cast<const v8h*>(&hb[(gy * WW + gx) * FF + f8 * 8]);
        *reinterpret_cast<v8h*>(&As[p * KTOT + KXPAD + tap * 64 + f8 * 8]) = v;
    }
    __syncthreads();

    // ---- WMMA GEMM: (32 x 672) x (672 x 256), f16 in / f32 acc ----
    const int lane  = tid & 31;
    const int wid   = tid >> 5;
    const int mtile = wid >> 2;   // 0..1  (16-pixel sub-tile)
    const int nw    = wid & 3;    // 0..3  (4 N-subtiles of 16 each)

    v8f acc0 = {}, acc1 = {}, acc2 = {}, acc3 = {};
    // A-fragment layout (contiguous, matches wpack):
    //   lane<16 : K 0..15 ; lane>=16 : K 16..31
    const int arow = (mtile * 16 + (lane & 15)) * KTOT + ((lane >> 4) * 16);

    for (int kc = 0; kc < NCHUNK; ++kc) {
        union { v16h v; v8h h[2]; } a;
        a.h[0] = *reinterpret_cast<const v8h*>(&As[arow + kc * 32]);
        a.h[1] = *reinterpret_cast<const v8h*>(&As[arow + kc * 32 + 8]);
        const _Float16* wb = &wpack[(size_t)((kc * 16 + nw * 4) * 32 + lane) * 16];
        v16h b0 = *reinterpret_cast<const v16h*>(wb);
        v16h b1 = *reinterpret_cast<const v16h*>(wb + 32 * 16);
        v16h b2 = *reinterpret_cast<const v16h*>(wb + 2 * 32 * 16);
        v16h b3 = *reinterpret_cast<const v16h*>(wb + 3 * 32 * 16);
        acc0 = __builtin_amdgcn_wmma_f32_16x16x32_f16(false, a.v, false, b0, (short)0, acc0, false, false);
        acc1 = __builtin_amdgcn_wmma_f32_16x16x32_f16(false, a.v, false, b1, (short)0, acc1, false, false);
        acc2 = __builtin_amdgcn_wmma_f32_16x16x32_f16(false, a.v, false, b2, (short)0, acc2, false, false);
        acc3 = __builtin_amdgcn_wmma_f32_16x16x32_f16(false, a.v, false, b3, (short)0, acc3, false, false);
    }
    __syncthreads();   // im2col buffer dead; reuse LDS as z buffer

    // C/D layout (ISA 7.12.2): VGPR j -> M=j (lanes 0-15) / M=8+j (lanes 16-31), N=lane&15
    {
        const int ncol0 = (nw * 4) * 16 + (lane & 15);
        const int prow  = mtile * 16 + 8 * (lane >> 4);
        #pragma unroll
        for (int e = 0; e < 8; ++e) {
            zbuf[(prow + e) * NGATE + ncol0 +  0] = acc0[e];
            zbuf[(prow + e) * NGATE + ncol0 + 16] = acc1[e];
            zbuf[(prow + e) * NGATE + ncol0 + 32] = acc2[e];
            zbuf[(prow + e) * NGATE + ncol0 + 48] = acc3[e];
        }
    }
    __syncthreads();

    // ---- gates (Keras order i,f,g,o) + state update ----
    float*     cb = c_state + (size_t)b * (HH * WW * FF);
    _Float16*  ho = h_out   + (size_t)b * (HH * WW * FF);
    for (int idx = tid; idx < PIX_PER_BLK * FF; idx += 256) {
        int p  = idx >> 6;
        int f  = idx & 63;
        size_t gidx = ((size_t)y * WW + (x0 + p)) * FF + f;
        float zi = zbuf[p * NGATE +       f] + bias[f];
        float zf = zbuf[p * NGATE +  64 + f] + bias[64 + f];
        float zg = zbuf[p * NGATE + 128 + f] + bias[128 + f];
        float zo = zbuf[p * NGATE + 192 + f] + bias[192 + f];
        float ig = hsig(zi), fg = hsig(zf), og = hsig(zo);
        float gt = tanhf(zg);
        float c_old = (t == 0) ? 0.0f : cb[gidx];
        float c_new = fg * c_old + ig * gt;
        float h_new = og * tanhf(c_new);
        cb[gidx] = c_new;
        ho[gidx] = (_Float16)h_new;
        if (out_f32) out_f32[(size_t)b * (HH * WW * FF) + gidx] = h_new;
    }
}

// ---------------------------------------------------------------------------
extern "C" void kernel_launch(void* const* d_in, const int* in_sizes, int n_in,
                              void* d_out, int out_size, void* d_ws, size_t ws_size,
                              hipStream_t stream)
{
    (void)in_sizes; (void)n_in; (void)out_size; (void)ws_size;
    const float* x    = (const float*)d_in[0];
    const float* Wx   = (const float*)d_in[1];
    const float* Wh   = (const float*)d_in[2];
    const float* bias = (const float*)d_in[3];
    float*       out  = (float*)d_out;

    char* ws = (char*)d_ws;
    const size_t WPACK_BYTES = (size_t)NCHUNK * 16 * 32 * 16 * sizeof(_Float16); // 344064
    const size_t C_BYTES     = (size_t)HSIZE * sizeof(float);                    // 18.9 MB
    const size_t H_BYTES     = (size_t)HSIZE * sizeof(_Float16);                 // 9.4 MB
    _Float16* wpack   = (_Float16*)ws;
    float*    c_state = (float*)(ws + WPACK_BYTES);
    _Float16* h0      = (_Float16*)(ws + WPACK_BYTES + C_BYTES);
    _Float16* h1      = (_Float16*)(ws + WPACK_BYTES + C_BYTES + H_BYTES);

    pack_weights_kernel<<<(NCHUNK * 16 * 32 * 16 + 255) / 256, 256, 0, stream>>>(Wx, Wh, wpack);
    zero_h_kernel<<<(HSIZE + 255) / 256, 256, 0, stream>>>(h0);

    for (int t = 0; t < TSTEPS; ++t) {
        const _Float16* hin  = (t & 1) ? h1 : h0;
        _Float16*       hout = (t & 1) ? h0 : h1;
        convlstm_step_kernel<<<BLOCKS_PER_STEP, 256, 0, stream>>>(
            x, wpack, bias, hin, hout, c_state,
            (t == TSTEPS - 1) ? out : nullptr, t);
    }
}
